// Block_39779987095924
// MI455X (gfx1250) — matrix operand order repaired and checked by reading, not compiled, not run on hardware
//
#include <hip/hip_runtime.h>
#include <hip/hip_bf16.h>
#include <cstdint>

// Problem constants (match reference)
#define BB   4
#define TT   4096
#define DD   256
#define HH   64
#define DSS  16
#define NP   (BB * TT)   // 16384 pixels (B * H * W), NHWC pixel index p = b*4096 + hh*64 + ww

typedef __attribute__((ext_vector_type(16))) __bf16       v16bf;
typedef __attribute__((ext_vector_type(8)))  float        v8f;
typedef __attribute__((ext_vector_type(8)))  unsigned int v8u;

// ---------------------------------------------------------------------------
// fp32 -> bf16 weight conversion
// ---------------------------------------------------------------------------
__global__ void k_f32_to_bf16(const float* __restrict__ src, __bf16* __restrict__ dst, int n) {
    int i = blockIdx.x * blockDim.x + threadIdx.x;
    if (i < n) dst[i] = (__bf16)src[i];
}

// ---------------------------------------------------------------------------
// LayerNorm over D=256 per row; also copies input to the shortcut output.
// One block (256 threads) per (b,t) row.
// ---------------------------------------------------------------------------
__global__ void k_layernorm(const float* __restrict__ x,
                            const float* __restrict__ gamma,
                            const float* __restrict__ beta,
                            float* __restrict__ shortcut,
                            __bf16* __restrict__ hln) {
    const int rowid = blockIdx.x;
    const int tid   = threadIdx.x;
    const size_t base = (size_t)rowid * DD;
    float v = x[base + tid];
    shortcut[base + tid] = v;

    float a = v, bsq = v * v;
    #pragma unroll
    for (int o = 16; o; o >>= 1) {
        a   += __shfl_down(a,   o, 32);
        bsq += __shfl_down(bsq, o, 32);
    }
    __shared__ float s1[8], s2[8];
    if ((tid & 31) == 0) { s1[tid >> 5] = a; s2[tid >> 5] = bsq; }
    __syncthreads();
    __shared__ float mu, rstd;
    if (tid == 0) {
        float sa = 0.f, sb = 0.f;
        #pragma unroll
        for (int i = 0; i < 8; ++i) { sa += s1[i]; sb += s2[i]; }
        float m   = sa * (1.0f / DD);
        float var = sb * (1.0f / DD) - m * m;
        mu = m;
        rstd = rsqrtf(var + 1e-5f);
    }
    __syncthreads();
    float h = (v - mu) * rstd * gamma[tid] + beta[tid];
    hln[base + tid] = (__bf16)h;
}

// ---------------------------------------------------------------------------
// WMMA bf16 GEMM: C[M,N] = A[M,K] * W[N,K]^T  (A, W bf16 row-major; C fp32)
// Each wave computes a 16 x (16*NT) tile using v_wmma_f32_16x16x32_bf16.
// Fragment layouts per CDNA5 ISA 7.12.2 (wave32).
// ---------------------------------------------------------------------------
template<int NT>
__global__ void k_gemm_bf16(const __bf16* __restrict__ A,
                            const __bf16* __restrict__ W,
                            float* __restrict__ C,
                            int M, int N, int K) {
    const int lane = threadIdx.x & 31;
    const int wid  = blockIdx.x * (blockDim.x >> 5) + (threadIdx.x >> 5);
    const int tilesN = N / (16 * NT);
    const int tilesM = M / 16;
    if (wid >= tilesM * tilesN) return;
    const int m0  = (wid / tilesN) * 16;
    const int n0  = (wid % tilesN) * (16 * NT);
    const int row  = lane & 15;
    const int half = lane >> 4;

    v8f acc[NT];
    const v8f vzero = {0.f, 0.f, 0.f, 0.f, 0.f, 0.f, 0.f, 0.f};
    #pragma unroll
    for (int t = 0; t < NT; ++t) acc[t] = vzero;

    const __bf16* aRow = A + (size_t)(m0 + row) * K;
    for (int k0 = 0; k0 < K; k0 += 32) {
        // A fragment: VGPR j holds K pair  (j>>2)*16 + half*8 + (j&3)*2
        v8u au;
        #pragma unroll
        for (int j = 0; j < 8; ++j) {
            int kk = k0 + ((j >> 2) << 4) + (half << 3) + ((j & 3) << 1);
            au[j] = *reinterpret_cast<const unsigned int*>(aRow + kk);
        }
        v16bf a = __builtin_bit_cast(v16bf, au);
        #pragma unroll
        for (int t = 0; t < NT; ++t) {
            // B fragment: lane = column n; VGPR j holds K pair (j>>2)*16 + half*8 + (j&3)*2
            const __bf16* wRow = W + (size_t)(n0 + t * 16 + row) * K;
            v8u bu;
            #pragma unroll
            for (int j = 0; j < 8; ++j) {
                int kk = k0 + ((j >> 2) << 4) + (half << 3) + ((j & 3) << 1);
                bu[j] = *reinterpret_cast<const unsigned int*>(wRow + kk);
            }
            v16bf b = __builtin_bit_cast(v16bf, bu);
            acc[t] = __builtin_amdgcn_wmma_f32_16x16x32_bf16(
                false, a, false, b, (short)0, acc[t], false, false);
        }
    }
    // C store: VGPR r -> M = m0 + half*8 + r, N = n0 + t*16 + row
    #pragma unroll
    for (int t = 0; t < NT; ++t) {
        #pragma unroll
        for (int r = 0; r < 8; ++r) {
            C[(size_t)(m0 + (half << 3) + r) * N + (n0 + t * 16 + row)] = acc[t][r];
        }
    }
}

// ---------------------------------------------------------------------------
// Depthwise 7x7 conv + bias, NHWC. Writes fp32 and bf16 copies.
// One block per pixel, 256 threads = channels (coalesced neighbor reads).
// ---------------------------------------------------------------------------
__global__ void k_dwconv7(const float* __restrict__ x0,
                          const float* __restrict__ w,     // [D,1,7,7]
                          const float* __restrict__ bias,
                          float* __restrict__ x1f,
                          __bf16* __restrict__ x1h) {
    const int p = blockIdx.x, d = threadIdx.x;
    const int b = p >> 12, t = p & 4095, hh = t >> 6, ww = t & 63;
    float acc = bias[d];
    for (int dy = 0; dy < 7; ++dy) {
        int y = hh + dy - 3; if ((unsigned)y >= 64u) continue;
        for (int dx = 0; dx < 7; ++dx) {
            int x = ww + dx - 3; if ((unsigned)x >= 64u) continue;
            acc = fmaf(w[d * 49 + dy * 7 + dx],
                       x0[((((size_t)b << 12) + ((size_t)y << 6) + x) << 8) + d], acc);
        }
    }
    x1f[((size_t)p << 8) + d] = acc;
    x1h[((size_t)p << 8) + d] = (__bf16)acc;
}

// ---------------------------------------------------------------------------
// Fused: gate projections (DS=16 dots) + sigmoid/normalize + chunk-2 scan +
// h*U + x*D + m_w direction merge. Exploits that CHUNK=2 makes the scan only
// 2 steps deep: col 2k is L*x, col 2k+1 adds gated row-neighbors of col 2k.
// Block = 256 channels, grid = (W/2 pairs, H rows, B).
// ---------------------------------------------------------------------------
__device__ __forceinline__ float dot16(const float* __restrict__ a, const float* b) {
    float s = 0.f;
    #pragma unroll
    for (int i = 0; i < 16; ++i) s = fmaf(a[i], b[i], s);
    return s;
}
__device__ __forceinline__ float sigmoidf(float x) { return 1.f / (1.f + __expf(-x)); }

__global__ void k_spn_fused(const float* __restrict__ x1,    // NHWC [p,256]
                            const float* __restrict__ xp,    // [p,16]
                            const float* __restrict__ wup,   // [3072,16]
                            const float* __restrict__ lup,   // [1024,16]
                            const float* __restrict__ uup,   // [1024,16]
                            const float* __restrict__ dup,   // [1024,16]
                            const float* __restrict__ mw,    // [4]
                            __bf16* __restrict__ merged) {   // NHWC [p,256] bf16
    const int k  = blockIdx.x;   // column pair
    const int hh = blockIdx.y;   // row
    const int b  = blockIdx.z;
    const int d  = threadIdx.x;
    const int w0 = 2 * k, w1 = w0 + 1;

    // Stage the 4 needed xp vectors: (hh-1,w0) (hh,w0) (hh+1,w0) (hh,w1)
    __shared__ float xpS[4][16];
    if (threadIdx.x < 64) {
        int pos = threadIdx.x >> 4, c = threadIdx.x & 15;
        int yy = (pos == 0) ? hh - 1 : ((pos == 2) ? hh + 1 : hh);
        int xx = (pos == 3) ? w1 : w0;
        float v = 0.f;
        if ((unsigned)yy < 64u)
            v = xp[((((size_t)b << 12) + ((size_t)yy << 6) + xx) << 4) + c];
        xpS[pos][c] = v;
    }
    __syncthreads();

    float m0acc = 0.f, m1acc = 0.f;
    #pragma unroll
    for (int f = 0; f < 4; ++f) {
        const int c = f * DD + d;    // channel in [0, 4D)
        // xs = {x, x^T, flip_w(x), flip_w(x^T)} lookup into NHWC x1; OOB row -> 0 (pad)
        auto xsld = [&](int yy, int xx) -> float {
            if ((unsigned)yy >= 64u) return 0.f;
            int Y, X;
            switch (f) {
                case 0:  Y = yy;       X = xx;       break;
                case 1:  Y = xx;       X = yy;       break;
                case 2:  Y = yy;       X = 63 - xx;  break;
                default: Y = 63 - xx;  X = yy;       break;
            }
            return x1[((((size_t)b << 12) + ((size_t)Y << 6) + X) << 8) + d];
        };
        float xs_m1 = xsld(hh - 1, w0);
        float xs_0  = xsld(hh,     w0);
        float xs_p1 = xsld(hh + 1, w0);
        float xs_1  = xsld(hh,     w1);

        const float* lrow = lup + (size_t)c * 16;
        float h_m1 = dot16(lrow, xpS[0]) * xs_m1;   // h(col w0) at row hh-1
        float h_0  = dot16(lrow, xpS[1]) * xs_0;    // h(col w0) at row hh
        float h_p1 = dot16(lrow, xpS[2]) * xs_p1;   // h(col w0) at row hh+1

        // Gates at (hh, w1): Gl row c, Gm row 4D+c, Gr row 8D+c
        float gl = sigmoidf(dot16(wup + (size_t)c * 16,          xpS[3]));
        float gm = sigmoidf(dot16(wup + (size_t)(1024 + c) * 16, xpS[3]));
        float gr = sigmoidf(dot16(wup + (size_t)(2048 + c) * 16, xpS[3]));
        float s = (hh == 0) ? (gm + gr) : ((hh == 63) ? (gl + gm) : (gl + gm + gr));
        float inv = 1.f / fmaxf(s, 1e-7f);

        float h1 = dot16(lrow, xpS[3]) * xs_1 + (gl * h_m1 + gm * h_0 + gr * h_p1) * inv;

        const float* urow = uup + (size_t)c * 16;
        const float* drow = dup + (size_t)c * 16;
        float out0 = h_0 * dot16(urow, xpS[1]) + xs_0 * dot16(drow, xpS[1]);
        float out1 = h1  * dot16(urow, xpS[3]) + xs_1 * dot16(drow, xpS[3]);
        float w_f = mw[f];
        m0acc = fmaf(w_f, out0, m0acc);
        m1acc = fmaf(w_f, out1, m1acc);
    }
    size_t p0 = ((size_t)b << 12) + ((size_t)hh << 6) + w0;
    merged[(p0 << 8) + d]       = (__bf16)m0acc;
    merged[((p0 + 1) << 8) + d] = (__bf16)m1acc;
}

// ---------------------------------------------------------------------------
// Depthwise 3x3 conv (no bias) + x*relu(x), NHWC, writes bf16 for final GEMM.
// ---------------------------------------------------------------------------
__global__ void k_dwconv3_sqrelu(const float* __restrict__ y0,
                                 const float* __restrict__ w,   // [D,1,3,3]
                                 __bf16* __restrict__ y2) {
    const int p = blockIdx.x, d = threadIdx.x;
    const int b = p >> 12, t = p & 4095, hh = t >> 6, ww = t & 63;
    float acc = 0.f;
    #pragma unroll
    for (int dy = 0; dy < 3; ++dy) {
        int y = hh + dy - 1; if ((unsigned)y >= 64u) continue;
        #pragma unroll
        for (int dx = 0; dx < 3; ++dx) {
            int x = ww + dx - 1; if ((unsigned)x >= 64u) continue;
            acc = fmaf(w[d * 9 + dy * 3 + dx],
                       y0[((((size_t)b << 12) + ((size_t)y << 6) + x) << 8) + d], acc);
        }
    }
    float v = acc * fmaxf(acc, 0.f);
    y2[((size_t)p << 8) + d] = (__bf16)v;
}

// ---------------------------------------------------------------------------
// Launcher
// ---------------------------------------------------------------------------
extern "C" void kernel_launch(void* const* d_in, const int* in_sizes, int n_in,
                              void* d_out, int out_size, void* d_ws, size_t ws_size,
                              hipStream_t stream) {
    (void)in_sizes; (void)n_in; (void)out_size; (void)ws_size;
    const float* hidden     = (const float*)d_in[0];
    const float* in_proj_w  = (const float*)d_in[1];
    const float* conv7_w    = (const float*)d_in[2];
    const float* conv7_b    = (const float*)d_in[3];
    const float* xdown_w    = (const float*)d_in[4];
    const float* wup_w      = (const float*)d_in[5];
    const float* lup_w      = (const float*)d_in[6];
    const float* uup_w      = (const float*)d_in[7];
    const float* dup_w      = (const float*)d_in[8];
    const float* m_w        = (const float*)d_in[9];
    const float* norm_w     = (const float*)d_in[10];
    const float* norm_b     = (const float*)d_in[11];
    const float* outconv_w  = (const float*)d_in[12];
    const float* outdconv_w = (const float*)d_in[13];
    const float* outproj_w  = (const float*)d_in[14];

    float* out      = (float*)d_out;
    float* shortcut = out + (size_t)NP * DD;   // second tuple element

    // Workspace carve-out (~82 MB)
    char*  w   = (char*)d_ws;
    size_t off = 0;
    auto alloc = [&](size_t bytes) -> void* {
        void* p = w + off;
        off = (off + bytes + 255) & ~(size_t)255;
        return p;
    };
    __bf16* hln  = (__bf16*)alloc((size_t)NP * DD * 2);
    float*  x0   = (float*) alloc((size_t)NP * DD * 4);
    float*  x1f  = (float*) alloc((size_t)NP * DD * 4);
    __bf16* x1h  = (__bf16*)alloc((size_t)NP * DD * 2);
    float*  xp   = (float*) alloc((size_t)NP * DSS * 4);
    __bf16* mrg  = (__bf16*)alloc((size_t)NP * DD * 2);
    float*  y0   = (float*) alloc((size_t)NP * DD * 4);
    __bf16* y2   = (__bf16*)alloc((size_t)NP * DD * 2);
    __bf16* wInP = (__bf16*)alloc((size_t)DD * DD * 2);
    __bf16* wXd  = (__bf16*)alloc((size_t)DSS * DD * 2);
    __bf16* wOc  = (__bf16*)alloc((size_t)DD * DD * 2);
    __bf16* wOp  = (__bf16*)alloc((size_t)DD * DD * 2);

    // Weight conversions (tiny)
    k_f32_to_bf16<<<(DD * DD + 255) / 256, 256, 0, stream>>>(in_proj_w, wInP, DD * DD);
    k_f32_to_bf16<<<(DSS * DD + 255) / 256, 256, 0, stream>>>(xdown_w, wXd, DSS * DD);
    k_f32_to_bf16<<<(DD * DD + 255) / 256, 256, 0, stream>>>(outconv_w, wOc, DD * DD);
    k_f32_to_bf16<<<(DD * DD + 255) / 256, 256, 0, stream>>>(outproj_w, wOp, DD * DD);

    // LayerNorm + shortcut copy
    k_layernorm<<<NP, 256, 0, stream>>>(hidden, norm_w, norm_b, shortcut, hln);

    // in_proj: [16384,256] x [256,256]^T  (WMMA)
    {
        int waves = (NP / 16) * (DD / 64);
        k_gemm_bf16<4><<<(waves * 32 + 255) / 256, 256, 0, stream>>>(hln, wInP, x0, NP, DD, DD);
    }
    // depthwise 7x7 + bias
    k_dwconv7<<<NP, 256, 0, stream>>>(x0, conv7_w, conv7_b, x1f, x1h);

    // xdown: [16384,16] = x1 x [16,256]^T  (WMMA, N=16)
    {
        int waves = (NP / 16) * (DSS / 16);
        k_gemm_bf16<1><<<(waves * 32 + 255) / 256, 256, 0, stream>>>(x1h, wXd, xp, NP, DSS, DD);
    }

    // Fused gates + chunk-2 scan + merge
    {
        dim3 grid(HH / 2, HH, BB);   // (32 column pairs, 64 rows, 4 batches)
        k_spn_fused<<<grid, 256, 0, stream>>>(x1f, xp, wup_w, lup_w, uup_w, dup_w, m_w, mrg);
    }

    // outconv (WMMA)
    {
        int waves = (NP / 16) * (DD / 64);
        k_gemm_bf16<4><<<(waves * 32 + 255) / 256, 256, 0, stream>>>(mrg, wOc, y0, NP, DD, DD);
    }
    // depthwise 3x3 + x*relu(x)
    k_dwconv3_sqrelu<<<NP, 256, 0, stream>>>(y0, outdconv_w, y2);

    // outproj (WMMA) -> first tuple element of d_out
    {
        int waves = (NP / 16) * (DD / 64);
        k_gemm_bf16<4><<<(waves * 32 + 255) / 256, 256, 0, stream>>>(y2, wOp, out, NP, DD, DD);
    }
}
